// GAE_36928128811025
// MI455X (gfx1250) — compile-verified, matrix operand order checked by reading, not run on hardware
//
#include <hip/hip_runtime.h>
#include <math.h>

// ---------------- problem constants ----------------
constexpr int NUc = 2048;
constexpr int NVc = 4096;
constexpr int RRc = 5;
constexpr int Fc  = 512;
constexpr int H0c = 500;
constexpr int H0P = 512;   // padded
constexpr int H1c = 75;
constexpr int H1P = 96;    // padded to multiple of 32 for WMMA K
constexpr int Dc  = 128;
constexpr size_t PLANE = (size_t)NUc * NVc; // 8388608

// ---------------- vector types for WMMA ----------------
typedef __bf16 v16bf __attribute__((ext_vector_type(16)));
typedef float  v8f   __attribute__((ext_vector_type(8)));
typedef unsigned short ushort8 __attribute__((ext_vector_type(8)));
typedef float  float8 __attribute__((ext_vector_type(8)));

union Frag { v16bf v; ushort8 h[2]; };

__device__ __forceinline__ unsigned short f2bf(float f) {
    unsigned int u = __float_as_uint(f);
    unsigned int r = u + 0x7FFFu + ((u >> 16) & 1u);   // round-to-nearest-even
    return (unsigned short)(r >> 16);
}

// ---------------- 0: zero accumulators / degrees ----------------
__global__ void k_init(float* acc, float* di, float* du) {
    int i = blockIdx.x * 256 + threadIdx.x;
    if (i < 4)   acc[i] = 0.0f;
    if (i < NUc) di[i]  = 0.0f;
    if (i < NVc) du[i]  = 0.0f;
}

// ---------------- 1: m -> mT(bf16), degree sums ----------------
__global__ void k_prep_m(const float* __restrict__ m, unsigned short* __restrict__ mT,
                         float* __restrict__ di, float* __restrict__ du) {
    __shared__ float tile[32][33];
    const int r  = blockIdx.z;
    const int u0 = blockIdx.y * 32;
    const int v0 = blockIdx.x * 32;
    const int tx = threadIdx.x;   // 0..31
    const int ty = threadIdx.y;   // 0..7
    const size_t mbase = ((size_t)r * NUc + u0) * (size_t)NVc + v0;
    for (int i = 0; i < 4; ++i) {
        int ul = ty + 8 * i;
        tile[ul][tx] = m[mbase + (size_t)ul * NVc + tx];
    }
    __syncthreads();
    const int tid = ty * 32 + tx;
    {   // transposed bf16 writes (coalesced along u)
        const int wu = tid & 31, wv = tid >> 5;
        for (int i = 0; i < 4; ++i) {
            int vl = wv + 8 * i;
            mT[((size_t)r * NVc + v0 + vl) * (size_t)NUc + u0 + wu] = f2bf(tile[wu][vl]);
        }
    }
    if (tid < 32) {                   // row (user) partial sums -> di
        float s = 0.0f;
        for (int vv = 0; vv < 32; ++vv) s += tile[tid][vv];
        atomicAdd(&di[u0 + tid], s);
    } else if (tid < 64) {            // col (item) partial sums -> du
        const int c = tid - 32;
        float s = 0.0f;
        for (int uu = 0; uu < 32; ++uu) s += tile[uu][c];
        atomicAdd(&du[v0 + c], s);
    }
}

// ---------------- 2: cumsum(gcl_w*) -> W^T bf16, padded ----------------
__global__ void k_prep_w(const float* __restrict__ gu, const float* __restrict__ gv,
                         unsigned short* __restrict__ WuT, unsigned short* __restrict__ WvT) {
    const int idx = blockIdx.x * 256 + threadIdx.x;   // 0 .. 512*512-1
    const float* src = blockIdx.y ? gv : gu;
    unsigned short* dst = blockIdx.y ? WvT : WuT;
    const int f = idx & 511, hcol = idx >> 9;
    float s = 0.0f;
    for (int r = 0; r < RRc; ++r) {
        if (hcol < H0c) s += src[((size_t)r * Fc + f) * H0c + hcol];
        dst[((size_t)r * H0P + hcol) * (size_t)Fc + f] = f2bf(hcol < H0c ? s : 0.0f);
    }
}

// ---------------- 3: gather features (bf16) + relu side ----------------
__global__ void k_gather(const int* __restrict__ uidx, const int* __restrict__ vidx,
                         const float* __restrict__ ufeat, const float* __restrict__ vfeat,
                         const float* __restrict__ uemb, const float* __restrict__ vemb,
                         unsigned short* __restrict__ uf, unsigned short* __restrict__ vf,
                         float* __restrict__ side) {
    const int pos = blockIdx.x * 256 + threadIdx.x;
    const int isV = blockIdx.y;
    const int M   = isV ? NVc : NUc;
    if (pos >= M * (Fc + Dc)) return;
    const int* idx = isV ? vidx : uidx;
    if (pos < M * Fc) {
        const int row = pos >> 9, ff = pos & 511;
        const float* feat = isV ? vfeat : ufeat;
        unsigned short* o = isV ? vf : uf;
        o[pos] = f2bf(feat[(size_t)idx[row] * Fc + ff]);
    } else {
        const int p = pos - M * Fc;
        const int row = p >> 7, dd = p & 127;
        const float* emb = isV ? vemb : uemb;
        const int so = isV ? NUc : 0;
        side[(size_t)(so + row) * Dc + dd] = fmaxf(emb[(size_t)idx[row] * Dc + dd], 0.0f);
    }
}

// ---------------- 4: hu/hv = X(bf16) @ W[r], output TRANSPOSED bf16 ----------------
// X: (M x 512) bf16 row-major. WT: (R x 512 x 512) bf16 (B^T, row = h, contiguous k=f).
// YT: (R x 512 x M) bf16: YT[r][h][m].  (operands are L2-resident; simple tiling OK)
__global__ void k_gemm_feat(const unsigned short* __restrict__ X,
                            const unsigned short* __restrict__ WT,
                            unsigned short* __restrict__ YT, int M) {
    const int r    = blockIdx.z;
    const int lane = threadIdx.x & 31;
    const int wave = threadIdx.x >> 5;
    const int m0   = blockIdx.x * 64 + wave * 16;
    const int n0   = blockIdx.y * 16;
    const int l15  = lane & 15, hh = lane >> 4;
    const unsigned short* arow = X + (size_t)(m0 + l15) * Fc;
    const unsigned short* brow = WT + ((size_t)r * H0P + n0 + l15) * (size_t)Fc;
    v8f acc = {};
    for (int k0 = 0; k0 < Fc; k0 += 32) {
        Frag a, b;
        a.h[0] = *(const ushort8*)(arow + k0 + hh * 8);
        a.h[1] = *(const ushort8*)(arow + k0 + 16 + hh * 8);
        b.h[0] = *(const ushort8*)(brow + k0 + hh * 8);
        b.h[1] = *(const ushort8*)(brow + k0 + 16 + hh * 8);
        acc = __builtin_amdgcn_wmma_f32_16x16x32_bf16(false, a.v, false, b.v,
                                                      (short)0, acc, false, false);
    }
    // D: element e -> (m = e + 8*hh, n = l15); transposed store, contiguous in m
    unsigned short* out = YT + (size_t)r * H0P * M + (size_t)(n0 + l15) * M + m0 + hh * 8;
    ushort8 w;
    for (int e = 0; e < 8; ++e) w[e] = f2bf(acc[e]);
    *(ushort8*)out = w;
}

// ---------------- 5: z = relu( (1/deg) * sum_r A_r @ B_r + 5*bias ) ----------------
// A: f32 (aF32=1, converted on the fly) or bf16; row-major (M x K) per segment.
// BT: bf16 (R x 512 x K): row = output col h, contiguous k.
// Streaming-A-optimized: grid (M/16, 2). Block = 4 waves sharing the same 16 A rows
// (A fragment loads dedup in WGP L0); each wave holds 4 accumulators covering 64
// n-columns, so a block covers 256 columns and A streams from HBM only twice total.
__global__ void k_gemm_z(const void* __restrict__ A, int aF32, size_t aSeg,
                         const unsigned short* __restrict__ BT, size_t bSeg,
                         const float* __restrict__ deg, const float* __restrict__ bias,
                         float* __restrict__ zOut, int M, int K) {
    const int lane = threadIdx.x & 31, wave = threadIdx.x >> 5;
    const int m0    = blockIdx.x * 16;
    const int nbase = blockIdx.y * 256 + wave * 64;   // this wave's first col (4 tiles)
    const int l15 = lane & 15, hh = lane >> 4;

    v8f acc[4] = {v8f{}, v8f{}, v8f{}, v8f{}};
    for (int r = 0; r < RRc; ++r) {
        const unsigned short* brow0 = BT + (size_t)r * bSeg + (size_t)(nbase + l15) * K;
        if (aF32) {
            const float* arow = (const float*)A + (size_t)r * aSeg + (size_t)(m0 + l15) * K;
            for (int k0 = 0; k0 < K; k0 += 32) {
                Frag a;
                float8 f0 = *(const float8*)(arow + k0 + hh * 8);
                float8 f1 = *(const float8*)(arow + k0 + 16 + hh * 8);
                ushort8 c0, c1;
                for (int e = 0; e < 8; ++e) { c0[e] = f2bf(f0[e]); c1[e] = f2bf(f1[e]); }
                a.h[0] = c0; a.h[1] = c1;
                for (int j = 0; j < 4; ++j) {
                    Frag b;
                    const unsigned short* brow = brow0 + (size_t)(j * 16) * K;
                    b.h[0] = *(const ushort8*)(brow + k0 + hh * 8);
                    b.h[1] = *(const ushort8*)(brow + k0 + 16 + hh * 8);
                    acc[j] = __builtin_amdgcn_wmma_f32_16x16x32_bf16(
                        false, a.v, false, b.v, (short)0, acc[j], false, false);
                }
            }
        } else {
            const unsigned short* arow =
                (const unsigned short*)A + (size_t)r * aSeg + (size_t)(m0 + l15) * K;
            for (int k0 = 0; k0 < K; k0 += 32) {
                Frag a;
                a.h[0] = *(const ushort8*)(arow + k0 + hh * 8);
                a.h[1] = *(const ushort8*)(arow + k0 + 16 + hh * 8);
                for (int j = 0; j < 4; ++j) {
                    Frag b;
                    const unsigned short* brow = brow0 + (size_t)(j * 16) * K;
                    b.h[0] = *(const ushort8*)(brow + k0 + hh * 8);
                    b.h[1] = *(const ushort8*)(brow + k0 + 16 + hh * 8);
                    acc[j] = __builtin_amdgcn_wmma_f32_16x16x32_bf16(
                        false, a.v, false, b.v, (short)0, acc[j], false, false);
                }
            }
        }
    }
    // epilogue: per-row 1/deg scale, +5*bias, relu
    float cinv[8];
    for (int e = 0; e < 8; ++e) {
        const float dg = deg[m0 + hh * 8 + e];
        cinv[e] = dg > 0.0f ? 1.0f / dg : 0.0f;
    }
    for (int j = 0; j < 4; ++j) {
        const int col = nbase + j * 16 + l15;
        const float bv = (col < H0c) ? 5.0f * bias[col] : 0.0f;
        for (int e = 0; e < 8; ++e) {
            const int row = m0 + hh * 8 + e;
            zOut[(size_t)row * H0P + col] = fmaxf(cinv[e] * acc[j][e] + bv, 0.0f);
        }
    }
}

// ---------------- 6: hidden = sigmoid(z @ d1w + side @ d2w) ----------------
__global__ void k_hidden(const float* __restrict__ z, const float* __restrict__ side,
                         const float* __restrict__ d1w, const float* __restrict__ d2w,
                         float* __restrict__ uh, unsigned short* __restrict__ vhb) {
    __shared__ float zr[H0c];
    __shared__ float sr[Dc];
    const int row = blockIdx.x;
    const int tid = threadIdx.x;   // 128
    for (int i = tid; i < H0c; i += 128) zr[i] = z[(size_t)row * H0P + i];
    if (tid < Dc) sr[tid] = side[(size_t)row * Dc + tid];
    __syncthreads();
    if (tid < H1P) {
        float hval = 0.0f;
        if (tid < H1c) {
            float s = 0.0f;
            for (int i = 0; i < H0c; ++i) s += zr[i] * d1w[i * H1c + tid];
            for (int i = 0; i < Dc;  ++i) s += sr[i] * d2w[i * H1c + tid];
            hval = 1.0f / (1.0f + __expf(-s));
        }
        if (row < NUc) { if (tid < H1c) uh[(size_t)row * H1c + tid] = hval; }
        else            vhb[(size_t)(row - NUc) * H1P + tid] = f2bf(hval);   // zero pad cols
    }
}

// ---------------- 7: uQ[r][u][:] = uh[u] @ (a[r,0]P0 + a[r,1]P1), bf16 padded ----------------
__global__ void k_uq(const float* __restrict__ uh, const float* __restrict__ P,
                     const float* __restrict__ a_coef, unsigned short* __restrict__ uQ) {
    __shared__ float ur[H1c];
    const int u = blockIdx.x, r = blockIdx.y;
    const int j = threadIdx.x;   // 96
    if (j < H1c) ur[j] = uh[(size_t)u * H1c + j];
    __syncthreads();
    unsigned short o = 0;
    if (j < H1c) {
        const float a0 = a_coef[r * 2 + 0], a1 = a_coef[r * 2 + 1];
        float s = 0.0f;
        for (int i = 0; i < H1c; ++i) {
            float q = a0 * P[i * H1c + j] + a1 * P[H1c * H1c + i * H1c + j];
            s += ur[i] * q;
        }
        o = f2bf(s);
    }
    uQ[((size_t)r * NUc + u) * H1P + j] = o;
}

// ---------------- 8: logits -> log_softmax -> m_hat / loss / rmse partials ----------------
__global__ void k_final(const unsigned short* __restrict__ uQ,
                        const unsigned short* __restrict__ vhb,
                        const float* __restrict__ t,
                        float* __restrict__ out, float* __restrict__ acc) {
    const int tid  = threadIdx.x;
    const int lane = tid & 31, wave = tid >> 5;
    const int u0 = blockIdx.x * 16;
    const int v0 = blockIdx.y * 64 + wave * 16;
    const int l15 = lane & 15, hh = lane >> 4;

    // B fragments from vh (B = vh^T, so BT rows == vh rows), reused across all r
    Frag b[3];
    const unsigned short* brow = vhb + (size_t)(v0 + l15) * H1P;
    for (int kc = 0; kc < 3; ++kc) {
        b[kc].h[0] = *(const ushort8*)(brow + kc * 32 + hh * 8);
        b[kc].h[1] = *(const ushort8*)(brow + kc * 32 + 16 + hh * 8);
    }
    v8f lr[RRc];
    for (int r = 0; r < RRc; ++r) {
        const unsigned short* arow = uQ + ((size_t)r * NUc + u0 + l15) * H1P;
        v8f a8 = {};
        for (int kc = 0; kc < 3; ++kc) {
            Frag a;
            a.h[0] = *(const ushort8*)(arow + kc * 32 + hh * 8);
            a.h[1] = *(const ushort8*)(arow + kc * 32 + 16 + hh * 8);
            a8 = __builtin_amdgcn_wmma_f32_16x16x32_bf16(false, a.v, false, b[kc].v,
                                                         (short)0, a8, false, false);
        }
        lr[r] = a8;
    }
    float lloss = 0.0f, lsq = 0.0f, lobs = 0.0f;
    const int v = v0 + l15;
    for (int e = 0; e < 8; ++e) {
        const int u = u0 + hh * 8 + e;
        const float l0 = lr[0][e], l1 = lr[1][e], l2 = lr[2][e], l3 = lr[3][e], l4 = lr[4][e];
        const float mx = fmaxf(fmaxf(fmaxf(l0, l1), fmaxf(l2, l3)), l4);
        const float e0 = __expf(l0 - mx), e1 = __expf(l1 - mx), e2 = __expf(l2 - mx),
                    e3 = __expf(l3 - mx), e4 = __expf(l4 - mx);
        const float ssum = e0 + e1 + e2 + e3 + e4;
        const float inv  = 1.0f / ssum;
        const float lse  = mx + __logf(ssum);
        const float mhat = (1.0f*e0 + 2.0f*e1 + 3.0f*e2 + 4.0f*e3 + 5.0f*e4) * inv;
        const size_t base = (size_t)u * NVc + v;
        const float t0 = t[base], t1 = t[base + PLANE], t2 = t[base + 2*PLANE],
                    t3 = t[base + 3*PLANE], t4 = t[base + 4*PLANE];
        const float tsum  = t0 + t1 + t2 + t3 + t4;
        const float trueR = t0 + 2.0f*t1 + 3.0f*t2 + 4.0f*t3 + 5.0f*t4;
        lloss += (lse - l0)*t0 + (lse - l1)*t1 + (lse - l2)*t2 + (lse - l3)*t3 + (lse - l4)*t4;
        const float mask = (tsum > 0.0f) ? 1.0f : 0.0f;
        lobs += mask;
        const float dd = mhat - trueR;
        lsq += dd * dd * mask;
        out[base] = mhat;
    }
    __shared__ float red[3][128];
    red[0][tid] = lloss; red[1][tid] = lsq; red[2][tid] = lobs;
    __syncthreads();
    for (int s = 64; s > 0; s >>= 1) {
        if (tid < s) {
            red[0][tid] += red[0][tid + s];
            red[1][tid] += red[1][tid + s];
            red[2][tid] += red[2][tid + s];
        }
        __syncthreads();
    }
    if (tid == 0) {
        atomicAdd(&acc[0], red[0][0]);
        atomicAdd(&acc[1], red[1][0]);
        atomicAdd(&acc[2], red[2][0]);
    }
}

// ---------------- 9: finalize scalars ----------------
__global__ void k_finalize(const float* __restrict__ acc, float* __restrict__ out) {
    const float n = fmaxf(acc[2], 1.0f);
    out[PLANE]     = acc[0] / n;
    out[PLANE + 1] = sqrtf(acc[1] / n);
}

// ---------------- host launch ----------------
extern "C" void kernel_launch(void* const* d_in, const int* in_sizes, int n_in,
                              void* d_out, int out_size, void* d_ws, size_t ws_size,
                              hipStream_t stream) {
    const int*   u          = (const int*)d_in[0];
    const int*   v          = (const int*)d_in[1];
    const float* m          = (const float*)d_in[2];
    const float* t          = (const float*)d_in[3];
    const float* u_features = (const float*)d_in[4];
    const float* v_features = (const float*)d_in[5];
    const float* u_emb_w    = (const float*)d_in[6];
    const float* v_emb_w    = (const float*)d_in[7];
    const float* gcl_wu     = (const float*)d_in[8];
    const float* gcl_wv     = (const float*)d_in[9];
    const float* gcl_b      = (const float*)d_in[10];
    const float* dense1_w   = (const float*)d_in[11];
    const float* dense2_w   = (const float*)d_in[12];
    const float* P          = (const float*)d_in[13];
    const float* a_coef     = (const float*)d_in[14];
    float* out = (float*)d_out;

    char* w = (char*)d_ws;
    float* acc  = (float*)w; w += 4 * sizeof(float);
    float* di   = (float*)w; w += (size_t)NUc * 4;
    float* du   = (float*)w; w += (size_t)NVc * 4;
    float* z    = (float*)w; w += (size_t)(NUc + NVc) * H0P * 4;
    float* side = (float*)w; w += (size_t)(NUc + NVc) * Dc * 4;
    float* uh   = (float*)w; w += (size_t)NUc * H1c * 4;
    unsigned short* mT  = (unsigned short*)w; w += (size_t)RRc * NVc * NUc * 2;
    unsigned short* WuT = (unsigned short*)w; w += (size_t)RRc * H0P * Fc * 2;
    unsigned short* WvT = (unsigned short*)w; w += (size_t)RRc * H0P * Fc * 2;
    unsigned short* uf  = (unsigned short*)w; w += (size_t)NUc * Fc * 2;
    unsigned short* vf  = (unsigned short*)w; w += (size_t)NVc * Fc * 2;
    unsigned short* huT = (unsigned short*)w; w += (size_t)RRc * H0P * NUc * 2;
    unsigned short* hvT = (unsigned short*)w; w += (size_t)RRc * H0P * NVc * 2;
    unsigned short* vhb = (unsigned short*)w; w += (size_t)NVc * H1P * 2;
    unsigned short* uQ  = (unsigned short*)w; w += (size_t)RRc * NUc * H1P * 2;

    k_init<<<dim3(16), dim3(256), 0, stream>>>(acc, di, du);
    k_prep_m<<<dim3(NVc / 32, NUc / 32, RRc), dim3(32, 8), 0, stream>>>(m, mT, di, du);
    k_prep_w<<<dim3(1024, 2), dim3(256), 0, stream>>>(gcl_wu, gcl_wv, WuT, WvT);
    k_gather<<<dim3((NVc * (Fc + Dc) + 255) / 256, 2), dim3(256), 0, stream>>>(
        u, v, u_features, v_features, u_emb_w, v_emb_w, uf, vf, side);
    k_gemm_feat<<<dim3(NUc / 64, H0P / 16, RRc), dim3(128), 0, stream>>>(uf, WuT, huT, NUc);
    k_gemm_feat<<<dim3(NVc / 64, H0P / 16, RRc), dim3(128), 0, stream>>>(vf, WvT, hvT, NVc);
    // zu: A = m (f32, converted inline), B^T = hvT, deg = di (per user)
    k_gemm_z<<<dim3(NUc / 16, 2), dim3(128), 0, stream>>>(
        (const void*)m, 1, (size_t)NUc * NVc, hvT, (size_t)H0P * NVc, di, gcl_b,
        z, NUc, NVc);
    // zv: A = mT (bf16), B^T = huT, deg = du (per item)
    k_gemm_z<<<dim3(NVc / 16, 2), dim3(128), 0, stream>>>(
        (const void*)mT, 0, (size_t)NVc * NUc, huT, (size_t)H0P * NUc, du, gcl_b,
        z + (size_t)NUc * H0P, NVc, NUc);
    k_hidden<<<dim3(NUc + NVc), dim3(128), 0, stream>>>(z, side, dense1_w, dense2_w, uh, vhb);
    k_uq<<<dim3(NUc, RRc), dim3(96), 0, stream>>>(uh, P, a_coef, uQ);
    k_final<<<dim3(NUc / 16, NVc / 64), dim3(128), 0, stream>>>(uQ, vhb, t, out, acc);
    k_finalize<<<dim3(1), dim3(1), 0, stream>>>(acc, out);
}
